// NTKViTSelfAttention_50586124812753
// MI455X (gfx1250) — matrix operand order repaired and checked by reading, not compiled, not run on hardware
//
#include <hip/hip_runtime.h>

#define B_   16
#define S_   1024
#define HID_ 768
#define H_   12
#define DH_  64

typedef __bf16 bf16;
typedef __attribute__((ext_vector_type(16))) __bf16 v16bf;
typedef __attribute__((ext_vector_type(8)))  __bf16 v8bf;
typedef __attribute__((ext_vector_type(8)))  float  v8f;
typedef __attribute__((ext_vector_type(4)))  int    v4i;

// ---- CDNA5 async / TDM feature detection (compile-safe) --------------------
#if defined(__has_builtin)
#  if __has_builtin(__builtin_amdgcn_global_load_async_to_lds_b128)
#    define HAVE_ASYNC 1
#  endif
#  if __has_builtin(__builtin_amdgcn_tensor_load_to_lds) && (__clang_major__ <= 22)
#    define HAVE_TDM 1
#  endif
#endif
#ifndef HAVE_ASYNC
#  define HAVE_ASYNC 0
#endif
#ifndef HAVE_TDM
#  define HAVE_TDM 0
#endif

#if HAVE_ASYNC
typedef __attribute__((address_space(1))) v4i* g4_ptr;  // global int4*
typedef __attribute__((address_space(3))) v4i* l4_ptr;  // LDS int4*
#define TO_GLOBAL(p) ((g4_ptr)(uintptr_t)(p))
#define TO_LDS(p)    ((l4_ptr)(unsigned int)(uintptr_t)(p))
__device__ __forceinline__ void wait_async0() {
#if __has_builtin(__builtin_amdgcn_s_wait_asynccnt)
  __builtin_amdgcn_s_wait_asynccnt(0);
#else
  asm volatile("s_wait_asynccnt 0x0" ::: "memory");
#endif
}
#endif

#if HAVE_TDM
typedef unsigned int u32x4 __attribute__((ext_vector_type(4)));
typedef int i32x8 __attribute__((ext_vector_type(8)));
typedef int i32x4 __attribute__((ext_vector_type(4)));
__device__ __forceinline__ void wait_tensor0() {
#if __has_builtin(__builtin_amdgcn_s_wait_tensorcnt)
  __builtin_amdgcn_s_wait_tensorcnt(0);
#else
  asm volatile("s_wait_tensorcnt 0x0" ::: "memory");
#endif
}
#endif

__device__ __forceinline__ v8f wmma_bf16(v16bf a, v16bf b, v8f c) {
  // 8 args: (neg_a, A, neg_b, B, c_mod, C, reuse_a, reuse_b)
  return __builtin_amdgcn_wmma_f32_16x16x32_bf16(false, a, false, b, (short)0, c,
                                                 false, false);
}

// A-matrix fragment (16x32 bf16): lane L (L<16) holds row M=L, K in {0..7,16..23};
// lanes 16..31 hold same rows with K in {8..15,24..31}.
__device__ __forceinline__ v16bf frag_a(const bf16* row, int lane) {
  const v8bf* p = (const v8bf*)row;
  int ks = lane >> 4;
  v8bf lo = p[ks];
  v8bf hi = p[ks + 2];
  v16bf r;
#pragma unroll
  for (int i = 0; i < 8; ++i) { r[i] = lo[i]; r[i + 8] = hi[i]; }
  return r;
}

// B-matrix fragment (32x16 bf16): lane L holds column N=L%16; lanes 0..15 hold
// K in {0..7,16..23}, lanes 16..31 hold K in {8..15,24..31}.
__device__ __forceinline__ v16bf frag_b(const bf16* col, int lane) {
  const v8bf* p = (const v8bf*)col;
  int ks = lane >> 4;
  v8bf lo = p[ks];
  v8bf hi = p[ks + 2];
  v16bf r;
#pragma unroll
  for (int i = 0; i < 8; ++i) { r[i] = lo[i]; r[i + 8] = hi[i]; }
  return r;
}

// ---------------------------------------------------------------------------
// Kernel 1: C[m,n] = sum_k X[m,k]*W[n,k] + bias[n]  (NT GEMM, bf16 WMMA)
// ---------------------------------------------------------------------------
#define GM 128
#define GN 64
#define GK 32
#define LDA 40

__global__ __launch_bounds__(256) void qkv_gemm_kernel(
    const float* __restrict__ X, const float* __restrict__ W,
    const float* __restrict__ bias, bf16* __restrict__ outq) {
  __shared__ bf16 As[GM][LDA];
  __shared__ bf16 Bs[GN][LDA];
  const int tid  = threadIdx.x;
  const int lane = tid & 31;
  const int wid  = tid >> 5;
  const int m0   = blockIdx.x * GM;
  const int n0   = blockIdx.y * GN;
  const int wm   = (wid & 3) * 32;
  const int wn   = (wid >> 2) * 32;

  v8f acc[2][2];
#pragma unroll
  for (int i = 0; i < 2; ++i)
#pragma unroll
    for (int j = 0; j < 2; ++j) {
      v8f z = {0, 0, 0, 0, 0, 0, 0, 0};
      acc[i][j] = z;
    }

  for (int k0 = 0; k0 < HID_; k0 += GK) {
    {  // stage A tile (f32 -> bf16): 128x32
      int r = tid >> 1, c = (tid & 1) * 16;
      const float* src = X + (size_t)(m0 + r) * HID_ + k0 + c;
#pragma unroll
      for (int i = 0; i < 16; i += 4) {
        float4 f = *(const float4*)(src + i);
        As[r][c + i]     = (bf16)f.x;
        As[r][c + i + 1] = (bf16)f.y;
        As[r][c + i + 2] = (bf16)f.z;
        As[r][c + i + 3] = (bf16)f.w;
      }
    }
    {  // stage B tile: 64x32
      int r = tid >> 2, c = (tid & 3) * 8;
      const float* src = W + (size_t)(n0 + r) * HID_ + k0 + c;
#pragma unroll
      for (int i = 0; i < 8; i += 4) {
        float4 f = *(const float4*)(src + i);
        Bs[r][c + i]     = (bf16)f.x;
        Bs[r][c + i + 1] = (bf16)f.y;
        Bs[r][c + i + 2] = (bf16)f.z;
        Bs[r][c + i + 3] = (bf16)f.w;
      }
    }
    __syncthreads();
    if (k0 + GK < HID_) {
      __builtin_prefetch(X + (size_t)(m0 + (tid >> 1)) * HID_ + k0 + GK, 0, 0);
      __builtin_prefetch(W + (size_t)(n0 + (tid >> 2)) * HID_ + k0 + GK, 0, 0);
    }
#pragma unroll
    for (int i = 0; i < 2; ++i) {
      v16bf a = frag_a(&As[wm + i * 16 + (lane & 15)][0], lane);
#pragma unroll
      for (int j = 0; j < 2; ++j) {
        v16bf b = frag_b(&Bs[wn + j * 16 + (lane & 15)][0], lane);
        acc[i][j] = wmma_bf16(a, b, acc[i][j]);
      }
    }
    __syncthreads();
  }

  const int hi8 = (lane >> 4) * 8;
#pragma unroll
  for (int i = 0; i < 2; ++i) {
#pragma unroll
    for (int j = 0; j < 2; ++j) {
      int n = n0 + wn + j * 16 + (lane & 15);
      int hh = n >> 6, dd = n & 63;
      float bval = bias[n];
#pragma unroll
      for (int r = 0; r < 8; ++r) {
        int m = m0 + wm + i * 16 + hi8 + r;
        int bi = m >> 10, si = m & 1023;
        outq[(((size_t)bi * H_ + hh) * S_ + si) * DH_ + dd] =
            (bf16)(acc[i][j][r] + bval);
      }
    }
  }
}

// ---------------------------------------------------------------------------
// Kernel 2: flash attention + phi correction, per (b, h, 64-query tile).
// ---------------------------------------------------------------------------
#define LDQ 72   // padded bf16 row stride (144B); pad = 4 DWORDs per 32 DWORDs

__global__ __launch_bounds__(128) void attn_kernel(
    const bf16* __restrict__ qb, const bf16* __restrict__ kb,
    const bf16* __restrict__ vb, const float* __restrict__ phi_k,
    const float* __restrict__ phi_kv, float* __restrict__ out) {
  __shared__ bf16 Qs[64][LDQ];
  __shared__ bf16 Ks[64][LDQ];
  __shared__ bf16 Vt[64][LDQ];
  __shared__ bf16 Ps[4][16][LDQ];
  __shared__ bf16 Ph[64][LDQ];
  __shared__ float Kv[64][DH_];
  __shared__ float Pk[64];

  const int tid  = threadIdx.x;
  const int lane = tid & 31;
  const int w    = tid >> 5;
  const int s0   = blockIdx.x * 64;
  const int h    = blockIdx.y;
  const int bz   = blockIdx.z;
  const size_t headBase = ((size_t)bz * H_ + h) * S_ * DH_;

  {  // ---- load Q tile into LDS (async copy engine if available)
    int r = tid >> 1, half = (tid & 1) * 32;
    const bf16* g = qb + headBase + (size_t)(s0 + r) * DH_ + half;
#if HAVE_ASYNC
#pragma unroll
    for (int i = 0; i < 4; ++i)
      __builtin_amdgcn_global_load_async_to_lds_b128(
          TO_GLOBAL(g + i * 8), TO_LDS(&Qs[r][half + i * 8]), 0, 0);
#else
    const v8bf* src = (const v8bf*)g;
#pragma unroll
    for (int i = 0; i < 4; ++i) *(v8bf*)&Qs[r][half + i * 8] = src[i];
#endif
  }
  if (tid < 64) Pk[tid] = fabsf(phi_k[h * DH_ + tid]);
  {  // phi_kv head slice
    const float* src = phi_kv + (size_t)h * DH_ * DH_;
#pragma unroll 8
    for (int i = 0; i < 32; ++i) {
      int idx = tid * 32 + i;
      Kv[idx >> 6][idx & 63] = src[idx];
    }
  }
#if HAVE_ASYNC
  wait_async0();
#endif
  __syncthreads();
  {  // phi_q = elu(q / DH^0.25) + 1
#pragma unroll 8
    for (int i = 0; i < 32; ++i) {
      int idx = tid * 32 + i;
      int r = idx >> 6, c = idx & 63;
      float x = (float)Qs[r][c] * 0.35355339059f;
      float p = (x > 0.0f ? x : __expf(x) - 1.0f) + 1.0f;
      Ph[r][c] = (bf16)p;
    }
  }
  __syncthreads();

  const int m0w = w * 16;
  v16bf aq0 = frag_a(&Qs[m0w + (lane & 15)][0], lane);
  v16bf aq1 = frag_a(&Qs[m0w + (lane & 15)][32], lane);

  v8f acc[4];
  float rm[8], rs[8];
#pragma unroll
  for (int j = 0; j < 4; ++j) {
    v8f z = {0, 0, 0, 0, 0, 0, 0, 0};
    acc[j] = z;
  }
#pragma unroll
  for (int r = 0; r < 8; ++r) { rm[r] = -1e30f; rs[r] = 0.0f; }

  const int hi8 = (lane >> 4) * 8;

  for (int t0 = 0; t0 < S_; t0 += 64) {
    // ---- stage K chunk: TDM > async > vector copy
#if HAVE_TDM
    if (w == 0) {
      unsigned long long ga =
          (unsigned long long)(uintptr_t)(kb + headBase + (size_t)t0 * DH_);
      unsigned int la = (unsigned int)(uintptr_t)&Ks[0][0];
      u32x4 g0;
      g0[0] = 1u;                                   // count=1 (valid descriptor)
      g0[1] = la;                                   // lds_addr
      g0[2] = (unsigned int)(ga & 0xFFFFFFFFu);     // global_addr[31:0]
      g0[3] = (unsigned int)((ga >> 32) & 0x1FFFFFFu) | (2u << 30);  // +type=2
      i32x8 g1;
      // data_size=2B(code1)<<16 | pad_enable<<20 | pad_interval(32dw=code4)<<22
      // | pad_amount(4dw=code3)<<25
      g1[0] = (1 << 16) | (1 << 20) | (4 << 22) | (3 << 25);
      g1[1] = (64 << 16);      // tensor_dim0 = 64 elems (bits 79:48)
      g1[2] = (1024 << 16);    // tensor_dim1 = 1024 rows (bits 111:80)
      g1[3] = (64 << 16);      // tile_dim0 = 64 (bits 127:112)
      g1[4] = 64;              // tile_dim1 = 64 (bits 143:128)
      g1[5] = 64;              // tensor_dim0_stride = 64 (bits 207:160)
      g1[6] = 0;
      g1[7] = 0;
      i32x4 z4 = {0, 0, 0, 0};
      __builtin_amdgcn_tensor_load_to_lds(g0, g1, z4, z4, 0);
      wait_tensor0();
    }
    {
      int r = tid >> 1, half = (tid & 1) * 32;
      const bf16* srcv = vb + headBase + (size_t)(t0 + r) * DH_ + half;
#pragma unroll 8
      for (int i = 0; i < 32; ++i) Vt[half + i][r] = srcv[i];
    }
#else
    {
      int r = tid >> 1, half = (tid & 1) * 32;
      const bf16* gk = kb + headBase + (size_t)(t0 + r) * DH_ + half;
#if HAVE_ASYNC
#pragma unroll
      for (int i = 0; i < 4; ++i)
        __builtin_amdgcn_global_load_async_to_lds_b128(
            TO_GLOBAL(gk + i * 8), TO_LDS(&Ks[r][half + i * 8]), 0, 0);
#else
      const v8bf* srck = (const v8bf*)gk;
#pragma unroll
      for (int i = 0; i < 4; ++i) *(v8bf*)&Ks[r][half + i * 8] = srck[i];
#endif
      const bf16* srcv = vb + headBase + (size_t)(t0 + r) * DH_ + half;
#pragma unroll 8
      for (int i = 0; i < 32; ++i) Vt[half + i][r] = srcv[i];
      if (t0 + 64 < S_) {
        __builtin_prefetch(kb + headBase + (size_t)(t0 + 64 + r) * DH_ + half, 0, 0);
        __builtin_prefetch(vb + headBase + (size_t)(t0 + 64 + r) * DH_ + half, 0, 0);
      }
    }
#if HAVE_ASYNC
    wait_async0();
#endif
#endif
    __syncthreads();

    // ---- S = (Q K^T) / sqrt(DH)
    v8f sc[4];
#pragma unroll
    for (int j = 0; j < 4; ++j) {
      v8f z = {0, 0, 0, 0, 0, 0, 0, 0};
      sc[j] = z;
    }
#pragma unroll
    for (int j = 0; j < 4; ++j) {
      v16bf b0 = frag_b(&Ks[j * 16 + (lane & 15)][0], lane);
      sc[j] = wmma_bf16(aq0, b0, sc[j]);
      v16bf b1 = frag_b(&Ks[j * 16 + (lane & 15)][32], lane);
      sc[j] = wmma_bf16(aq1, b1, sc[j]);
    }
#pragma unroll
    for (int j = 0; j < 4; ++j)
#pragma unroll
      for (int r = 0; r < 8; ++r) sc[j][r] *= 0.125f;

    // ---- online softmax
#pragma unroll
    for (int r = 0; r < 8; ++r) {
      float m = fmaxf(fmaxf(sc[0][r], sc[1][r]), fmaxf(sc[2][r], sc[3][r]));
      m = fmaxf(m, __shfl_xor(m, 1, 32));
      m = fmaxf(m, __shfl_xor(m, 2, 32));
      m = fmaxf(m, __shfl_xor(m, 4, 32));
      m = fmaxf(m, __shfl_xor(m, 8, 32));
      float nm    = fmaxf(rm[r], m);
      float alpha = __expf(rm[r] - nm);
      rm[r] = nm;
      rs[r] *= alpha;
#pragma unroll
      for (int j = 0; j < 4; ++j) acc[j][r] *= alpha;
      float ps = 0.0f;
#pragma unroll
      for (int j = 0; j < 4; ++j) {
        float p = __expf(sc[j][r] - nm);
        ps += p;
        Ps[w][hi8 + r][j * 16 + (lane & 15)] = (bf16)p;
      }
      rs[r] += ps;
    }

    // ---- acc += P @ V
#pragma unroll
    for (int kk = 0; kk < 2; ++kk) {
      v16bf ap = frag_a(&Ps[w][(lane & 15)][kk * 32], lane);
#pragma unroll
      for (int j = 0; j < 4; ++j) {
        v16bf bvf = frag_b(&Vt[j * 16 + (lane & 15)][kk * 32], lane);
        acc[j] = wmma_bf16(ap, bvf, acc[j]);
      }
    }
    __syncthreads();
  }

  // ---- cross-lane row sums
#pragma unroll
  for (int r = 0; r < 8; ++r) {
    float s = rs[r];
    s += __shfl_xor(s, 1, 32);
    s += __shfl_xor(s, 2, 32);
    s += __shfl_xor(s, 4, 32);
    s += __shfl_xor(s, 8, 32);
    rs[r] = s;
  }

  // ---- fold phi corrections at final max, divide, store f32
#pragma unroll
  for (int r = 0; r < 8; ++r) {
    int row = m0w + hi8 + r;
    float cq = 0.0f;
#pragma unroll 8
    for (int k = 0; k < 64; ++k) cq += (float)Ph[row][k] * Pk[k];
    float ie    = __expf(-rm[r]);
    float denom = rs[r] + cq * ie;
#pragma unroll
    for (int j = 0; j < 4; ++j) {
      int d = j * 16 + (lane & 15);
      float cc = 0.0f;
#pragma unroll 8
      for (int k = 0; k < 64; ++k) cc += (float)Ph[row][k] * Kv[k][d];
      float val = (acc[j][r] + cc * ie) / denom;
      out[((size_t)bz * S_ + (s0 + row)) * HID_ + h * DH_ + d] = val;
    }
  }
}

extern "C" void kernel_launch(void* const* d_in, const int* in_sizes, int n_in,
                              void* d_out, int out_size, void* d_ws, size_t ws_size,
                              hipStream_t stream) {
  (void)in_sizes; (void)n_in; (void)out_size; (void)ws_size;
  const float* hs      = (const float*)d_in[0];
  const float* Wq      = (const float*)d_in[1];
  const float* bq      = (const float*)d_in[2];
  const float* Wk      = (const float*)d_in[3];
  const float* bk      = (const float*)d_in[4];
  const float* Wv      = (const float*)d_in[5];
  const float* bvp     = (const float*)d_in[6];
  const float* phi_k   = (const float*)d_in[7];
  const float* phi_kv  = (const float*)d_in[8];
  float* out = (float*)d_out;

  const size_t QKVE = (size_t)B_ * H_ * S_ * DH_;
  bf16* qb = (bf16*)d_ws;
  bf16* kb = qb + QKVE;
  bf16* vb = kb + QKVE;

  dim3 gg((B_ * S_) / GM, HID_ / GN);
  qkv_gemm_kernel<<<gg, 256, 0, stream>>>(hs, Wq, bq, qb);
  qkv_gemm_kernel<<<gg, 256, 0, stream>>>(hs, Wk, bk, kb);
  qkv_gemm_kernel<<<gg, 256, 0, stream>>>(hs, Wv, bvp, vb);

  attn_kernel<<<dim3(S_ / 64, H_, B_), 128, 0, stream>>>(qb, kb, vb, phi_k,
                                                         phi_kv, out);
}
